// _JetMoeAttention_85787676770834
// MI455X (gfx1250) — matrix operand order repaired and checked by reading, not compiled, not run on hardware
//
#include <hip/hip_runtime.h>
#include <hip/hip_bf16.h>
#include <math.h>

typedef __attribute__((ext_vector_type(16))) _Float16 v16h;
typedef __attribute__((ext_vector_type(8)))  float    v8f;

#define S_LEN 2048
#define HSZ   2048
#define NE    8
#define TOPK  2
#define NKVH  16
#define HD    128
#define NQH   (TOPK * NKVH)   // 32
#define QSZ   (NKVH * HD)     // 2048
#define DHALF (HD / 2)        // 64

// ---------------------------------------------------------------------------
// WMMA fragment helpers (this file's contiguous K order, same for A and B, wave32).
// A (16x32 f16): row = lane%16; elem j    -> K = half*16+j
// B (32x16 f16): col = lane%16; elem j    -> K = half*16+j
// C/D (16x16 f32): elem r -> M = half*8+r, N = lane%16
// ---------------------------------------------------------------------------
static __device__ inline v16h frag_a_f32(const float* rowptr, int kbase, int half) {
  v16h a;
#pragma unroll
  for (int j = 0; j < 8; ++j) a[j]     = (_Float16)rowptr[kbase + half * 16 + j];
#pragma unroll
  for (int j = 0; j < 8; ++j) a[8 + j] = (_Float16)rowptr[kbase + half * 16 + 8 + j];
  return a;
}

static __device__ inline v16h frag_a_f16(const _Float16* rowptr, int kbase, int half) {
  v16h a;
#pragma unroll
  for (int j = 0; j < 8; ++j) a[j]     = rowptr[kbase + half * 16 + j];
#pragma unroll
  for (int j = 0; j < 8; ++j) a[8 + j] = rowptr[kbase + half * 16 + 8 + j];
  return a;
}

// B fragment where K runs contiguously in memory for this lane's column.
static __device__ inline v16h frag_b_f32(const float* colptr, int kbase, int half) {
  v16h b;
#pragma unroll
  for (int j = 0; j < 16; ++j) b[j] = (_Float16)colptr[kbase + half * 16 + j];
  return b;
}

static __device__ inline v16h frag_b_f16(const _Float16* colptr, int kbase, int half) {
  v16h b;
#pragma unroll
  for (int j = 0; j < 16; ++j) b[j] = colptr[kbase + half * 16 + j];
  return b;
}

static __device__ inline v8f wmma16(v16h a, v16h b, v8f c) {
  return __builtin_amdgcn_wmma_f32_16x16x32_f16(false, a, false, b, (short)0, c,
                                                false, false);
}

// ---------------------------------------------------------------------------
// K1: router — logits, top-2, softmax gates, bucket scatter
// ---------------------------------------------------------------------------
__global__ void router_kernel(const float* __restrict__ hidden,
                              const float* __restrict__ rw,
                              float* __restrict__ gates,
                              int* __restrict__ counts,
                              int* __restrict__ bucket) {
  int s = blockIdx.x * blockDim.x + threadIdx.x;
  if (s >= S_LEN) return;
  const float* hrow = hidden + (size_t)s * HSZ;
  float acc[NE];
#pragma unroll
  for (int e = 0; e < NE; ++e) acc[e] = 0.f;
  for (int k = 0; k < HSZ; ++k) {
    float x = hrow[k];
#pragma unroll
    for (int e = 0; e < NE; ++e) acc[e] += x * rw[e * HSZ + k];
  }
  int e0 = 0; float v0 = acc[0];
#pragma unroll
  for (int e = 1; e < NE; ++e) if (acc[e] > v0) { v0 = acc[e]; e0 = e; }
  int e1 = (e0 == 0) ? 1 : 0; float v1 = -3.0e38f;
#pragma unroll
  for (int e = 0; e < NE; ++e)
    if (e != e0 && acc[e] > v1) { v1 = acc[e]; e1 = e; }
  float g0 = 1.0f / (1.0f + expf(v1 - v0));
  gates[s * TOPK + 0] = g0;
  gates[s * TOPK + 1] = 1.0f - g0;
  int p0 = atomicAdd(&counts[e0], 1);
  bucket[e0 * S_LEN + p0] = (s << 1);
  int p1 = atomicAdd(&counts[e1], 1);
  bucket[e1 * S_LEN + p1] = (s << 1) | 1;
}

// ---------------------------------------------------------------------------
// K2: KV projection (+RoPE on K). 64-row x 16-col register-blocked tiles.
// grid: (S/64, 64 k-pair-tiles + 128 v-tiles)
// ---------------------------------------------------------------------------
__global__ __launch_bounds__(32) void kv_kernel(const float* __restrict__ hidden,
                                                const float* __restrict__ kv_w,
                                                const float* __restrict__ cosb,
                                                const float* __restrict__ sinb,
                                                _Float16* __restrict__ kbuf,
                                                _Float16* __restrict__ vbuf) {
  const int xt = blockIdx.x;   // 64-row block
  const int y  = blockIdx.y;
  const int lane = threadIdx.x, half = lane >> 4, nlo = lane & 15;
  const float* arow[4];
#pragma unroll
  for (int i = 0; i < 4; ++i)
    arow[i] = hidden + (size_t)(xt * 64 + i * 16 + nlo) * HSZ;

  if (y < 64) {  // K part: RoPE pair columns (d, d+64) of one kv head
    int kvh = y >> 2, dp = (y & 3) * 16;
    int o0 = kvh * HD + dp;
    const float* bc0 = kv_w + (size_t)(o0 + nlo) * HSZ;
    const float* bc1 = kv_w + (size_t)(o0 + 64 + nlo) * HSZ;
    v8f c0[4], c1[4];
    v8f zero = {};
#pragma unroll
    for (int i = 0; i < 4; ++i) { c0[i] = zero; c1[i] = zero; }
    for (int kc = 0; kc < HSZ / 32; ++kc) {
      int kbase = kc * 32;
      v16h b0 = frag_b_f32(bc0, kbase, half);
      v16h b1 = frag_b_f32(bc1, kbase, half);
      __builtin_prefetch(bc0 + kbase + 32, 0, 1);
      __builtin_prefetch(bc1 + kbase + 32, 0, 1);
#pragma unroll
      for (int i = 0; i < 4; ++i) {
        v16h a = frag_a_f32(arow[i], kbase, half);
        c0[i] = wmma16(a, b0, c0[i]);
        c1[i] = wmma16(a, b1, c1[i]);
      }
    }
#pragma unroll
    for (int i = 0; i < 4; ++i) {
#pragma unroll
      for (int r = 0; r < 8; ++r) {
        int srow = xt * 64 + i * 16 + half * 8 + r;
        int d2 = dp + nlo;
        float c  = cosb[(size_t)srow * DHALF + d2];
        float sn = sinb[(size_t)srow * DHALF + d2];
        float x1 = c0[i][r], x2 = c1[i][r];
        kbuf[(size_t)srow * QSZ + kvh * HD + d2]      = (_Float16)(x1 * c - x2 * sn);
        kbuf[(size_t)srow * QSZ + kvh * HD + d2 + 64] = (_Float16)(x1 * sn + x2 * c);
      }
    }
  } else {       // V part: plain GEMM tile
    int cbase = (y - 64) * 16;
    const float* bc = kv_w + (size_t)(QSZ + cbase + nlo) * HSZ;
    v8f c0[4];
    v8f zero = {};
#pragma unroll
    for (int i = 0; i < 4; ++i) c0[i] = zero;
    for (int kc = 0; kc < HSZ / 32; ++kc) {
      int kbase = kc * 32;
      v16h b = frag_b_f32(bc, kbase, half);
      __builtin_prefetch(bc + kbase + 32, 0, 1);
#pragma unroll
      for (int i = 0; i < 4; ++i)
        c0[i] = wmma16(frag_a_f32(arow[i], kbase, half), b, c0[i]);
    }
#pragma unroll
    for (int i = 0; i < 4; ++i) {
#pragma unroll
      for (int r = 0; r < 8; ++r) {
        int srow = xt * 64 + i * 16 + half * 8 + r;
        vbuf[(size_t)srow * QSZ + cbase + nlo] = (_Float16)c0[i][r];
      }
    }
  }
}

// ---------------------------------------------------------------------------
// K3: Q projection (grouped GEMM over expert buckets) + RoPE
// 64-row bucket super-tiles. grid: (E * S/64, 64 pair-tiles)
// ---------------------------------------------------------------------------
__global__ __launch_bounds__(32) void qproj_kernel(const float* __restrict__ hidden,
                                                   const float* __restrict__ q_w,
                                                   const float* __restrict__ cosb,
                                                   const float* __restrict__ sinb,
                                                   const int* __restrict__ counts,
                                                   const int* __restrict__ bucket,
                                                   _Float16* __restrict__ qbuf) {
  const int e    = blockIdx.x >> 5;
  const int tile = blockIdx.x & 31;
  const int n    = counts[e];
  const int base = tile * 64;
  if (base >= n) return;
  const int lane = threadIdx.x, half = lane >> 4, nlo = lane & 15;

  const float* arow[4];
#pragma unroll
  for (int i = 0; i < 4; ++i) {
    int idx = base + i * 16 + nlo;
    if (idx >= n) idx = n - 1;                        // pad with last row
    arow[i] = hidden + (size_t)(bucket[e * S_LEN + idx] >> 1) * HSZ;
  }

  int kvh = blockIdx.y >> 2, dp = (blockIdx.y & 3) * 16;
  int q0 = kvh * HD + dp;
  const float* wq  = q_w + (size_t)e * QSZ * HSZ;
  const float* bc0 = wq + (size_t)(q0 + nlo) * HSZ;
  const float* bc1 = wq + (size_t)(q0 + 64 + nlo) * HSZ;

  v8f c0[4], c1[4];
  v8f zero = {};
#pragma unroll
  for (int i = 0; i < 4; ++i) { c0[i] = zero; c1[i] = zero; }
  for (int kc = 0; kc < HSZ / 32; ++kc) {
    int kbase = kc * 32;
    v16h b0 = frag_b_f32(bc0, kbase, half);
    v16h b1 = frag_b_f32(bc1, kbase, half);
    __builtin_prefetch(bc0 + kbase + 32, 0, 1);
    __builtin_prefetch(bc1 + kbase + 32, 0, 1);
#pragma unroll
    for (int i = 0; i < 4; ++i) {
      v16h a = frag_a_f32(arow[i], kbase, half);
      c0[i] = wmma16(a, b0, c0[i]);
      c1[i] = wmma16(a, b1, c1[i]);
    }
  }
#pragma unroll
  for (int i = 0; i < 4; ++i) {
#pragma unroll
    for (int r = 0; r < 8; ++r) {
      int idx = base + i * 16 + half * 8 + r;
      if (idx >= n) continue;
      int entry = bucket[e * S_LEN + idx];
      int tok = entry >> 1, ks = entry & 1;
      int d2 = dp + nlo;
      float c  = cosb[(size_t)tok * DHALF + d2];
      float sn = sinb[(size_t)tok * DHALF + d2];
      float x1 = c0[i][r], x2 = c1[i][r];
      int hh = kvh * TOPK + ks;   // reference head order: h = kvh*K + kslot
      qbuf[((size_t)tok * NQH + hh) * HD + d2]      = (_Float16)(x1 * c - x2 * sn);
      qbuf[((size_t)tok * NQH + hh) * HD + d2 + 64] = (_Float16)(x1 * sn + x2 * c);
    }
  }
}

// ---------------------------------------------------------------------------
// K4: causal flash attention, one wave per (head, 16-row q tile)
// ---------------------------------------------------------------------------
__global__ __launch_bounds__(32) void attn_kernel(const _Float16* __restrict__ qbuf,
                                                  const _Float16* __restrict__ kbuf,
                                                  const _Float16* __restrict__ vbuf,
                                                  _Float16* __restrict__ attn_buf) {
  const int h   = blockIdx.x;          // 0..31
  const int t   = blockIdx.y;          // q tile
  const int kvh = h / TOPK;            // k_rep: head h -> kv head h//K
  const int lane = threadIdx.x, half = lane >> 4, nlo = lane & 15;
  __shared__ _Float16 Plds[16][32];

  const float scale = 0.08838834764831845f;  // 128^-0.5

  v16h qf[4];
  const _Float16* qrow = qbuf + ((size_t)(t * 16 + nlo) * NQH + h) * HD;
#pragma unroll
  for (int dc = 0; dc < 4; ++dc) qf[dc] = frag_a_f16(qrow, dc * 32, half);

  float m[8], l[8];
  v8f acc[8];
  v8f zero = {};
#pragma unroll
  for (int r = 0; r < 8; ++r) { m[r] = -3.0e38f; l[r] = 0.f; }
#pragma unroll
  for (int dc = 0; dc < 8; ++dc) acc[dc] = zero;

  const int nk32 = (t + 2) >> 1;       // 32-key blocks through the diagonal
  for (int kb = 0; kb < nk32; ++kb) {
    int k0 = kb * 32;
    v8f sc0 = {}, sc1 = {};
    const _Float16* kcol0 = kbuf + (size_t)(k0 + nlo) * QSZ + kvh * HD;
    const _Float16* kcol1 = kbuf + (size_t)(k0 + 16 + nlo) * QSZ + kvh * HD;
    __builtin_prefetch(kcol0 + 32 * QSZ, 0, 1);
#pragma unroll
    for (int dc = 0; dc < 4; ++dc) {
      sc0 = wmma16(qf[dc], frag_b_f16(kcol0, dc * 32, half), sc0);
      sc1 = wmma16(qf[dc], frag_b_f16(kcol1, dc * 32, half), sc1);
    }
    float p0[8], p1[8];
#pragma unroll
    for (int r = 0; r < 8; ++r) {
      int qg = t * 16 + half * 8 + r;
      float x0 = sc0[r] * scale + ((k0 + nlo)      <= qg ? 0.f : -1.0e9f);
      float x1 = sc1[r] * scale + ((k0 + 16 + nlo) <= qg ? 0.f : -1.0e9f);
      float mx = fmaxf(x0, x1);
#pragma unroll
      for (int msk = 1; msk < 16; msk <<= 1) mx = fmaxf(mx, __shfl_xor(mx, msk, 32));
      float mnew  = fmaxf(m[r], mx);
      float alpha = expf(m[r] - mnew);
      p0[r] = expf(x0 - mnew);
      p1[r] = expf(x1 - mnew);
      float ps = p0[r] + p1[r];
#pragma unroll
      for (int msk = 1; msk < 16; msk <<= 1) ps += __shfl_xor(ps, msk, 32);
      l[r] = l[r] * alpha + ps;
      m[r] = mnew;
#pragma unroll
      for (int dc = 0; dc < 8; ++dc) acc[dc][r] *= alpha;
    }
    // C-layout -> A-layout re-staging through LDS
#pragma unroll
    for (int r = 0; r < 8; ++r) {
      Plds[half * 8 + r][nlo]      = (_Float16)p0[r];
      Plds[half * 8 + r][16 + nlo] = (_Float16)p1[r];
    }
    __syncthreads();
    v16h pa;
#pragma unroll
    for (int j = 0; j < 8; ++j) pa[j]     = Plds[nlo][half * 16 + j];
#pragma unroll
    for (int j = 0; j < 8; ++j) pa[8 + j] = Plds[nlo][half * 16 + 8 + j];
    __syncthreads();
#pragma unroll
    for (int dc = 0; dc < 8; ++dc) {
      const _Float16* vbase = vbuf + (size_t)k0 * QSZ + kvh * HD + dc * 16 + nlo;
      v16h vb;
#pragma unroll
      for (int j = 0; j < 16; ++j) vb[j] = vbase[(size_t)(half * 16 + j) * QSZ];
      acc[dc] = wmma16(pa, vb, acc[dc]);
    }
  }
#pragma unroll
  for (int r = 0; r < 8; ++r) {
    float rinv = 1.0f / l[r];
    int srow = t * 16 + half * 8 + r;
    _Float16* orow = attn_buf + ((size_t)srow * NQH + h) * HD;
#pragma unroll
    for (int dc = 0; dc < 8; ++dc)
      orow[dc * 16 + nlo] = (_Float16)(acc[dc][r] * rinv);
  }
}

// ---------------------------------------------------------------------------
// K5: output projection (grouped GEMM over buckets), gate applied at store
// 64-row bucket super-tiles. grid: (E * S/64, H/16 col tiles)
// ---------------------------------------------------------------------------
__global__ __launch_bounds__(32) void oproj_kernel(const _Float16* __restrict__ attn_buf,
                                                   const float* __restrict__ o_w,
                                                   const int* __restrict__ counts,
                                                   const int* __restrict__ bucket,
                                                   const float* __restrict__ gates,
                                                   float* __restrict__ contrib) {
  const int e    = blockIdx.x >> 5;
  const int tile = blockIdx.x & 31;
  const int n    = counts[e];
  const int base = tile * 64;
  if (base >= n) return;
  const int lane = threadIdx.x, half = lane >> 4, nlo = lane & 15;

  int tokA[4], ksA[4];
#pragma unroll
  for (int i = 0; i < 4; ++i) {
    int idx = base + i * 16 + nlo;
    if (idx >= n) idx = n - 1;
    int ent = bucket[e * S_LEN + idx];
    tokA[i] = ent >> 1; ksA[i] = ent & 1;
  }
  int h0 = blockIdx.y * 16;
  const float* bc = o_w + ((size_t)e * HSZ + h0 + nlo) * QSZ;

  v8f c[4];
  v8f zero = {};
#pragma unroll
  for (int i = 0; i < 4; ++i) c[i] = zero;
  for (int kc = 0; kc < QSZ / 32; ++kc) {
    int kbase = kc * 32;
    v16h b = frag_b_f32(bc, kbase, half);
    __builtin_prefetch(bc + kbase + 32, 0, 1);
#pragma unroll
    for (int i = 0; i < 4; ++i) {
      // attn4[s, kslot, q] with q = kvh*D + d  ->  attn_buf[s, kvh*K + kslot, d]
      const _Float16* arow =
          attn_buf + ((size_t)tokA[i] * NQH + (kbase / HD) * TOPK + ksA[i]) * HD +
          (kbase % HD);
      c[i] = wmma16(frag_a_f16(arow, 0, half), b, c[i]);
    }
  }
#pragma unroll
  for (int i = 0; i < 4; ++i) {
#pragma unroll
    for (int r = 0; r < 8; ++r) {
      int idx = base + i * 16 + half * 8 + r;
      if (idx >= n) continue;
      int entry = bucket[e * S_LEN + idx];
      int tok = entry >> 1, ks = entry & 1;
      float g = gates[tok * TOPK + ks];
      contrib[((size_t)tok * TOPK + ks) * HSZ + h0 + nlo] = g * c[i][r];
    }
  }
}

// ---------------------------------------------------------------------------
// K6: combine: out = bias + contrib[slot0] + contrib[slot1]
// ---------------------------------------------------------------------------
__global__ void combine_kernel(const float* __restrict__ contrib,
                               const float* __restrict__ bias,
                               float* __restrict__ out) {
  size_t i = (size_t)blockIdx.x * blockDim.x + threadIdx.x;
  if (i >= (size_t)S_LEN * HSZ) return;
  int hh = (int)(i % HSZ);
  size_t s = i / HSZ;
  out[i] = bias[hh] + contrib[(s * TOPK) * HSZ + hh] + contrib[(s * TOPK + 1) * HSZ + hh];
}

// ---------------------------------------------------------------------------
extern "C" void kernel_launch(void* const* d_in, const int* in_sizes, int n_in,
                              void* d_out, int out_size, void* d_ws, size_t ws_size,
                              hipStream_t stream) {
  const float* hidden = (const float*)d_in[0];
  // d_in[1] = attention_bias (causal -1e9 mask) — applied analytically in-kernel
  const float* cosb = (const float*)d_in[2];
  const float* sinb = (const float*)d_in[3];
  const float* rw   = (const float*)d_in[4];
  const float* q_w  = (const float*)d_in[5];
  const float* o_w  = (const float*)d_in[6];
  const float* kv_w = (const float*)d_in[7];
  const float* bias = (const float*)d_in[8];
  float* out = (float*)d_out;

  char* ws = (char*)d_ws;
  size_t off = 0;
  auto walloc = [&](size_t bytes) -> void* {
    void* p = ws + off;
    off = (off + bytes + 255) & ~(size_t)255;
    return p;
  };
  int*      counts  = (int*)walloc(NE * sizeof(int));
  int*      bucket  = (int*)walloc((size_t)NE * S_LEN * sizeof(int));
  float*    gates   = (float*)walloc((size_t)S_LEN * TOPK * sizeof(float));
  _Float16* kbuf    = (_Float16*)walloc((size_t)S_LEN * QSZ * sizeof(_Float16));
  _Float16* vbuf    = (_Float16*)walloc((size_t)S_LEN * QSZ * sizeof(_Float16));
  _Float16* qbuf    = (_Float16*)walloc((size_t)S_LEN * NQH * HD * sizeof(_Float16));
  _Float16* attn    = (_Float16*)walloc((size_t)S_LEN * NQH * HD * sizeof(_Float16));
  float*    contrib = (float*)walloc((size_t)S_LEN * TOPK * HSZ * sizeof(float));
  (void)ws_size; (void)in_sizes; (void)n_in; (void)out_size;

  hipMemsetAsync(counts, 0, NE * sizeof(int), stream);
  router_kernel<<<dim3((S_LEN + 255) / 256), dim3(256), 0, stream>>>(
      hidden, rw, gates, counts, bucket);
  kv_kernel<<<dim3(S_LEN / 64, 192), dim3(32), 0, stream>>>(
      hidden, kv_w, cosb, sinb, kbuf, vbuf);
  qproj_kernel<<<dim3(NE * (S_LEN / 64), 64), dim3(32), 0, stream>>>(
      hidden, q_w, cosb, sinb, counts, bucket, qbuf);
  attn_kernel<<<dim3(NQH, S_LEN / 16), dim3(32), 0, stream>>>(
      qbuf, kbuf, vbuf, attn);
  oproj_kernel<<<dim3(NE * (S_LEN / 64), 128), dim3(32), 0, stream>>>(
      attn, o_w, counts, bucket, gates, contrib);
  combine_kernel<<<dim3((S_LEN * HSZ + 255) / 256), dim3(256), 0, stream>>>(
      contrib, bias, out);
}
